// LiquidNeuralNetwork_30064771072123
// MI455X (gfx1250) — matrix operand order repaired and checked by reading, not compiled, not run on hardware
//
#include <hip/hip_runtime.h>
#include <math.h>

#define INPUT_DIM 512
#define HIDDEN    1024
#define OUT_DIM   512
#define BATCH     64
#define SEQ       512

typedef __bf16 bf16;
typedef __attribute__((ext_vector_type(16))) __bf16 v16bf;
typedef __attribute__((ext_vector_type(8)))  __bf16 v8bf;
typedef __attribute__((ext_vector_type(8)))  float  v8f;

// ---------- fragment helpers (ISA 7.12.2 layouts, wave32) ----------

// A fragment: 16x32 bf16, row-major source, base points at (row0, k0).
// lanes 0-15: M=lane, K={0..7}u{16..23}; lanes 16-31: K={8..15}u{24..31}.
__device__ __forceinline__ v16bf load_a_bf16(const bf16* base, int ld, int lane) {
  const int m = lane & 15, half = lane >> 4;
  const bf16* p = base + (long)m * ld + half * 8;
  union { v16bf v; v8bf h[2]; } u;
  u.h[0] = *(const v8bf*)(p);
  u.h[1] = *(const v8bf*)(p + 16);
  return u.v;
}

// B fragment: 32x16 bf16 from row-major W[N][K], base points at (n0, k0).
// lanes 0-15: N=lane, K={0..7}u{16..23}; lanes 16-31: K={8..15}u{24..31}.
__device__ __forceinline__ v16bf load_b_bf16(const bf16* base, int ld, int lane) {
  const int n = lane & 15, half = lane >> 4;
  const bf16* p = base + (long)n * ld + half * 8;
  union { v16bf v; v8bf h[2]; } u;
  u.h[0] = *(const v8bf*)(p);
  u.h[1] = *(const v8bf*)(p + 16);
  return u.v;
}

// C/D 16x16 f32: VGPR i holds (M = 8*(lane/16)+i, N = lane%16).
__device__ __forceinline__ v8f load_c_bf16(const bf16* base, int ld, int lane) {
  const int n = lane & 15, mb = (lane >> 4) * 8;
  v8f c;
#pragma unroll
  for (int i = 0; i < 8; ++i) c[i] = (float)base[(long)(mb + i) * ld + n];
  return c;
}

__device__ __forceinline__ void store_c_f32(float* base, int ld, int lane, v8f c) {
  const int n = lane & 15, mb = (lane >> 4) * 8;
#pragma unroll
  for (int i = 0; i < 8; ++i) base[(long)(mb + i) * ld + n] = c[i];
}

__device__ __forceinline__ void store_c_bf16(bf16* base, int ld, int lane, v8f c) {
  const int n = lane & 15, mb = (lane >> 4) * 8;
#pragma unroll
  for (int i = 0; i < 8; ++i) base[(long)(mb + i) * ld + n] = (bf16)c[i];
}

__device__ __forceinline__ void store_c_tanh_bf16(bf16* base, int ld, int lane, v8f c) {
  const int n = lane & 15, mb = (lane >> 4) * 8;
#pragma unroll
  for (int i = 0; i < 8; ++i) base[(long)(mb + i) * ld + n] = (bf16)tanhf(c[i]);
}

__device__ __forceinline__ v8f wmma_bf16(v16bf a, v16bf b, v8f c) {
  return __builtin_amdgcn_wmma_f32_16x16x32_bf16(false, a, false, b, (short)0, c,
                                                 false, false);
}

// ---------- pipelined 16Mx32N GEMM inner loop, FULLY UNROLLED ----------
// No backedge -> no loop-carried phis -> no v_mov buffer rotation and no
// WMMA-hazard v_nops; every k0 is a constant -> all addresses are
// base + immediate IOFFSET (no per-iteration SALU/VALU address math).
template <int K, int LDA, int LDB>
__device__ __forceinline__ void gemm_loop(const bf16* __restrict__ ab,
                                          const bf16* __restrict__ bb,
                                          int lane, v8f acc[2]) {
  v16bf a0 = load_a_bf16(ab, LDA, lane);
  v16bf b0[2];
#pragma unroll
  for (int j = 0; j < 2; ++j)
    b0[j] = load_b_bf16(bb + (size_t)j * 16 * LDB, LDB, lane);

#pragma unroll
  for (int k0 = 0; k0 < K; k0 += 64) {
    // stage 1: fetch k0+32 into buffer 1, compute on buffer 0
    v16bf a1 = load_a_bf16(ab + k0 + 32, LDA, lane);
    v16bf b1[2];
#pragma unroll
    for (int j = 0; j < 2; ++j)
      b1[j] = load_b_bf16(bb + (size_t)j * 16 * LDB + k0 + 32, LDB, lane);
#pragma unroll
    for (int j = 0; j < 2; ++j) acc[j] = wmma_bf16(a0, b0[j], acc[j]);

    // stage 2: fetch k0+64 into buffer 0 (constant-folded away on last stage),
    // compute on buffer 1
    if (k0 + 64 < K) {
      a0 = load_a_bf16(ab + k0 + 64, LDA, lane);
#pragma unroll
      for (int j = 0; j < 2; ++j)
        b0[j] = load_b_bf16(bb + (size_t)j * 16 * LDB + k0 + 64, LDB, lane);
    }
#pragma unroll
    for (int j = 0; j < 2; ++j) acc[j] = wmma_bf16(a1, b1[j], acc[j]);
  }
}

// ---------- utility kernels ----------

__global__ void k_cvt_bf16(const float* __restrict__ in, bf16* __restrict__ out, int n) {
  for (int i = blockIdx.x * blockDim.x + threadIdx.x; i < n;
       i += gridDim.x * blockDim.x)
    out[i] = (bf16)in[i];                 // native v_cvt_pk_bf16_f32, RTNE
}

__global__ void k_zero_bf16(bf16* __restrict__ out, int n) {
  const bf16 z = __builtin_bit_cast(bf16, (unsigned short)0);
  for (int i = blockIdx.x * blockDim.x + threadIdx.x; i < n;
       i += gridDim.x * blockDim.x)
    out[i] = z;
}

// ---------- phase 1: XP[s*64+b][h] = x[b][s][:] . W_ih[h][:] + b_ih[h] + b_hh[h] ----------
// grid: (32768/64, 1024/64) = (512, 16); block = 256 (8 waves, each 16Mx32N).

__global__ __launch_bounds__(256) void k_xproj(const bf16* __restrict__ xb,
                                               const bf16* __restrict__ Wih,
                                               const float* __restrict__ bih,
                                               const float* __restrict__ bhh,
                                               bf16* __restrict__ XP) {
  const int lane = threadIdx.x & 31;
  const int wave = threadIdx.x >> 5;
  const int r0 = blockIdx.x * 64 + (wave & 3) * 16;       // XP row (= s*64 + b)
  const int n0 = blockIdx.y * 64 + (wave >> 2) * 32;      // hidden column base
  const int b0 = r0 & 63;
  const int s  = r0 >> 6;
  const bf16* ab = xb + ((size_t)b0 * SEQ + s) * INPUT_DIM;
  const bf16* bb = Wih + (size_t)n0 * INPUT_DIM;
  const int n = lane & 15;

  v8f acc[2];
#pragma unroll
  for (int j = 0; j < 2; ++j) {
    float bsum = bih[n0 + j * 16 + n] + bhh[n0 + j * 16 + n];
#pragma unroll
    for (int i = 0; i < 8; ++i) acc[j][i] = bsum;
  }

  gemm_loop<INPUT_DIM, SEQ * INPUT_DIM, INPUT_DIM>(ab, bb, lane, acc);

#pragma unroll
  for (int j = 0; j < 2; ++j)
    store_c_bf16(XP + (size_t)r0 * HIDDEN + n0 + j * 16, HIDDEN, lane, acc[j]);
}

// ---------- phase 2: h_out = tanh(xp_s + h_in @ W_hh^T) ----------
// grid: 16 (N macro-tiles of 64); block = 256 (8 waves, each 16Mx32N); K = 1024.
// C initialized from XP (D = A*B + C absorbs the add).

__global__ __launch_bounds__(256) void k_step(const bf16* __restrict__ xp_s,
                                              const bf16* __restrict__ Whh,
                                              const bf16* __restrict__ h_in,
                                              bf16* __restrict__ h_out) {
  const int lane = threadIdx.x & 31;
  const int wave = threadIdx.x >> 5;
  const int m0 = (wave & 3) * 16;                         // batch rows
  const int nb = blockIdx.x * 64 + (wave >> 2) * 32;      // hidden cols
  const bf16* ab = h_in + (size_t)m0 * HIDDEN;
  const bf16* bb = Whh + (size_t)nb * HIDDEN;

  v8f acc[2];
#pragma unroll
  for (int j = 0; j < 2; ++j)
    acc[j] = load_c_bf16(xp_s + (size_t)m0 * HIDDEN + nb + j * 16, HIDDEN, lane);

  gemm_loop<HIDDEN, HIDDEN, HIDDEN>(ab, bb, lane, acc);

#pragma unroll
  for (int j = 0; j < 2; ++j)
    store_c_tanh_bf16(h_out + (size_t)m0 * HIDDEN + nb + j * 16, HIDDEN, lane,
                      acc[j]);
}

// ---------- phase 3: out = h_final @ W_fc^T + b_fc  ([64 x 512], K=1024) ----------
// grid: 8 (N macro-tiles of 64); block = 256.

__global__ __launch_bounds__(256) void k_fc(const bf16* __restrict__ h_fin,
                                            const bf16* __restrict__ Wfc,
                                            const float* __restrict__ bfc,
                                            float* __restrict__ out) {
  const int lane = threadIdx.x & 31;
  const int wave = threadIdx.x >> 5;
  const int m0 = (wave & 3) * 16;
  const int nb = blockIdx.x * 64 + (wave >> 2) * 32;
  const int n = lane & 15;
  const bf16* ab = h_fin + (size_t)m0 * HIDDEN;
  const bf16* bb = Wfc + (size_t)nb * HIDDEN;

  v8f acc[2];
#pragma unroll
  for (int j = 0; j < 2; ++j) {
    float bv = bfc[nb + j * 16 + n];
#pragma unroll
    for (int i = 0; i < 8; ++i) acc[j][i] = bv;
  }

  gemm_loop<HIDDEN, HIDDEN, HIDDEN>(ab, bb, lane, acc);

#pragma unroll
  for (int j = 0; j < 2; ++j)
    store_c_f32(out + (size_t)m0 * OUT_DIM + nb + j * 16, OUT_DIM, lane, acc[j]);
}

// ---------- host ----------

extern "C" void kernel_launch(void* const* d_in, const int* in_sizes, int n_in,
                              void* d_out, int out_size, void* d_ws, size_t ws_size,
                              hipStream_t stream) {
  const float* x    = (const float*)d_in[0];
  const float* W_ih = (const float*)d_in[1];
  const float* W_hh = (const float*)d_in[2];
  const float* b_ih = (const float*)d_in[3];
  const float* b_hh = (const float*)d_in[4];
  const float* W_fc = (const float*)d_in[5];
  const float* b_fc = (const float*)d_in[6];
  float* out = (float*)d_out;

  // workspace layout (all offsets >=256B aligned); total ~100.3 MB
  char* ws = (char*)d_ws;
  bf16* Wih_bf = (bf16*)(ws + 0);                      // 1 MB
  bf16* Whh_bf = (bf16*)(ws + 1048576ull);             // 2 MB
  bf16* Wfc_bf = (bf16*)(ws + 3145728ull);             // 1 MB
  bf16* h0     = (bf16*)(ws + 4194304ull);             // 128 KB
  bf16* h1     = (bf16*)(ws + 4325376ull);             // 128 KB
  bf16* x_bf   = (bf16*)(ws + 4456448ull);             // 32 MB
  bf16* XP     = (bf16*)(ws + 38010880ull);            // 64 MB

  // bandwidth-bound pre-passes: fp32 -> bf16
  k_cvt_bf16<<<512, 256, 0, stream>>>(W_ih, Wih_bf, HIDDEN * INPUT_DIM);
  k_cvt_bf16<<<1024, 256, 0, stream>>>(W_hh, Whh_bf, HIDDEN * HIDDEN);
  k_cvt_bf16<<<512, 256, 0, stream>>>(W_fc, Wfc_bf, OUT_DIM * HIDDEN);
  k_cvt_bf16<<<4096, 256, 0, stream>>>(x, x_bf, BATCH * SEQ * INPUT_DIM);
  k_zero_bf16<<<64, 256, 0, stream>>>(h0, BATCH * HIDDEN);

  // phase 1: big input-projection GEMM (34.4 GFLOP, WMMA-bound)
  k_xproj<<<dim3(512, 16), 256, 0, stream>>>(x_bf, Wih_bf, b_ih, b_hh, XP);

  // phase 2: 512 sequential steps (graph-captured; W_hh bf16 L2-resident)
  bf16* hin = h0;
  bf16* hout = h1;
  for (int s = 0; s < SEQ; ++s) {
    k_step<<<16, 256, 0, stream>>>(XP + (size_t)s * BATCH * HIDDEN, Whh_bf, hin, hout);
    bf16* t = hin; hin = hout; hout = t;
  }

  // phase 3: final FC
  k_fc<<<8, 256, 0, stream>>>(hin, Wfc_bf, b_fc, out);
}
